// DUBLIDNet_42125039239896
// MI455X (gfx1250) — matrix operand order repaired and checked by reading, not compiled, not run on hardware
//
#include <hip/hip_runtime.h>

#ifndef __has_builtin
#define __has_builtin(x) 0
#endif

#define NB 4
#define CIN 3
#define CH 16
#define HV 256
#define WV 256
#define HK 45
#define NL 10
#define FS 384
#define FS2 (FS * FS)

typedef __attribute__((ext_vector_type(16))) _Float16 v16h;
typedef __attribute__((ext_vector_type(8))) _Float16 v8h;
typedef __attribute__((ext_vector_type(8))) float v8f;
typedef __attribute__((ext_vector_type(4))) unsigned int u32x4;
typedef __attribute__((ext_vector_type(4))) int i32x4;
typedef __attribute__((ext_vector_type(8))) int i32x8;

#define GS_LOOP(i, total)                                                     \
  for (long i = (long)blockIdx.x * blockDim.x + threadIdx.x; i < (total);     \
       i += (long)gridDim.x * blockDim.x)

// ---------- complex helpers (float2 = re,im) ----------
__device__ inline float2 cadd(float2 a, float2 b) { return make_float2(a.x + b.x, a.y + b.y); }
__device__ inline float2 csub(float2 a, float2 b) { return make_float2(a.x - b.x, a.y - b.y); }
__device__ inline float2 cmul(float2 a, float2 b) {
  return make_float2(a.x * b.x - a.y * b.y, a.x * b.y + a.y * b.x);
}
// conj(a)*b
__device__ inline float2 cmulcj(float2 a, float2 b) {
  return make_float2(a.x * b.x + a.y * b.y, a.x * b.y - a.y * b.x);
}
__device__ inline float2 cscale(float2 a, float s) { return make_float2(a.x * s, a.y * s); }
__device__ inline float mag2(float2 a) { return a.x * a.x + a.y * a.y; }
__device__ inline float softt(float v, float b) {
  float m = fabsf(v) - b;
  return m > 0.f ? copysignf(m, v) : 0.f;
}

// =====================================================================
// Weight prep: center w0 (subtract per-(co,ci) mean), build reflected f16
// copy for the first conv layer (zero-padded K 27->32), keep centered f32
// for the Fw0 spectra used by the Wiener solve.
// =====================================================================
__global__ void prep_w0(const float* __restrict__ w0, _Float16* __restrict__ w0rH,
                        float* __restrict__ w0c) {
  int t = threadIdx.x;
  if (t < 48) {  // (co, ci)
    int co = t / 3, ci = t % 3;
    const float* w = w0 + (co * 3 + ci) * 9;
    float m = 0.f;
    for (int j = 0; j < 9; ++j) m += w[j];
    m *= (1.f / 9.f);
    for (int j = 0; j < 9; ++j) {
      float v = w[j] - m;
      w0c[(co * 3 + ci) * 9 + j] = v;
      int ky = j / 3, kx = j % 3;
      // reflected (::-1,::-1) into im2col order k = ci*9 + ky*3 + kx
      w0rH[co * 32 + ci * 9 + (2 - ky) * 3 + (2 - kx)] = (_Float16)v;
    }
  }
  if (t < 16)
    for (int k = 27; k < 32; ++k) w0rH[t * 32 + k] = (_Float16)0.f;
}

// ws layers -> f16, layout [l][co][k], K=144 zero-padded to 160
__global__ void prep_ws(const float* __restrict__ ws, _Float16* __restrict__ wsH) {
  GS_LOOP(i, (long)(NL - 1) * CH * 160) {
    long l = i / (CH * 160);
    int rem = (int)(i % (CH * 160));
    int co = rem / 160, k = rem % 160;
    float v = 0.f;
    if (k < 144) v = ws[(l * CH + co) * 144 + k];
    wsH[i] = (_Float16)v;
  }
}

// =====================================================================
// 3x3 conv (cross-correlation, SAME) as implicit GEMM on WMMA:
//   D(16 x 16px) += A(16co x 32k) * B(32k x 16px), f16 in / f32 acc.
// Compile-time (Cin, Kpad) -> fully unrolled chunk loop; im2col staged in
// LDS as [pixel][k] so every A/B fragment is two contiguous ds_load_b128.
// =====================================================================
template <int CIN_T, int KPAD>
__global__ __launch_bounds__(256) void conv3x3_wmma(const float* __restrict__ in,
                                                    const _Float16* __restrict__ wgt,
                                                    float* __restrict__ out) {
  constexpr int KEFF = CIN_T * 9;
  __shared__ _Float16 wLds[16 * KPAD];
  __shared__ _Float16 raw[CIN_T * 3 * 132];
  __shared__ _Float16 imcol[128 * KPAD];
  int x0 = blockIdx.x * 128, y = blockIdx.y, nb = blockIdx.z;
  int tid = threadIdx.x;

  for (int i = tid; i < 16 * KPAD; i += 256) wLds[i] = wgt[i];
  for (int i = tid; i < CIN_T * 3 * 130; i += 256) {
    int ci = i / 390, rem = i - ci * 390;
    int row = rem / 130, col = rem - row * 130;
    int gy = y + row - 1, gx = x0 + col - 1;
    float v = 0.f;
    if ((unsigned)gy < (unsigned)HV && (unsigned)gx < (unsigned)WV)
      v = in[(((long)nb * CIN_T + ci) * HV + gy) * WV + gx];
    raw[(ci * 3 + row) * 132 + col] = (_Float16)v;
  }
  __syncthreads();

  // im2col expansion: imcol[px][k], k = ci*9 + ky*3 + kx, zero-padded K
  for (int e = tid; e < 128 * KPAD; e += 256) {
    int k = e >> 7, px = e & 127;
    _Float16 v = (_Float16)0.f;
    if (k < KEFF) {
      int ci = k / 9, t = k - ci * 9, ky = t / 3, kx = t - ky * 3;
      v = raw[(ci * 3 + ky) * 132 + px + kx];
    }
    imcol[px * KPAD + k] = v;
  }
  __syncthreads();

  int lane = tid & 31, wv = tid >> 5;
  int grp = lane >> 4, mn = lane & 15;
  const _Float16* aBase = &wLds[mn * KPAD];
  const _Float16* bBase = &imcol[(wv * 16 + mn) * KPAD];
  v8f acc = {};
#pragma unroll
  for (int c0 = 0; c0 < KPAD; c0 += 32) {
    // A (16-bit 16x32): lanes 0-15 hold K c0+0..7 & c0+16..23, lanes 16-31 +8
    v8h a0 = *(const v8h*)(aBase + c0 + grp * 8);
    v8h a1 = *(const v8h*)(aBase + c0 + 16 + grp * 8);
    // B (16-bit 32x16): lanes 0-15 hold K c0+0..7 & c0+16..23, lanes 16-31 +8
    v8h b0 = *(const v8h*)(bBase + c0 + grp * 8);
    v8h b1 = *(const v8h*)(bBase + c0 + 16 + grp * 8);
    v16h a = __builtin_shufflevector(a0, a1, 0, 1, 2, 3, 4, 5, 6, 7, 8, 9, 10, 11, 12, 13, 14, 15);
    v16h b = __builtin_shufflevector(b0, b1, 0, 1, 2, 3, 4, 5, 6, 7, 8, 9, 10, 11, 12, 13, 14, 15);
    acc = __builtin_amdgcn_wmma_f32_16x16x32_f16(false, a, false, b, (short)0, acc, false, false);
  }
  // D layout: lane holds n = lane&15; VGPR r holds m = r + 8*(lane>>4)
  int xc = x0 + wv * 16 + mn;
#pragma unroll
  for (int r = 0; r < 8; ++r) {
    int m = r + 8 * grp;
    out[(((long)nb * CH + m) * HV + y) * WV + xc] = acc[r];
  }
}

// =====================================================================
// 384-point complex FFT (384 = 3 * 128): three radix-2 DIT FFT-128s on
// decimated subsequences, then a radix-3 twiddle combine. One line per
// block, LDS resident. Contiguous lines are staged via the TDM.
// =====================================================================
__global__ __launch_bounds__(128) void fft384_kernel(float2* data, int nLines, int innerCount,
                                                     long outerStride, long innerStride,
                                                     long elemStride, int inverse) {
  __shared__ float2 S[FS];
  __shared__ float2 B[FS];
  long line = blockIdx.x;
  if (line >= nLines) return;
  long base = (line / innerCount) * outerStride + (line % innerCount) * innerStride;
  float2* p = data + base;
  int tid = threadIdx.x;

  bool staged = false;
#if defined(__gfx1250__) && __has_builtin(__builtin_amdgcn_tensor_load_to_lds) && \
    __has_builtin(__builtin_amdgcn_s_wait_tensorcnt)
  if (elemStride == 1) {
    if (tid < 32) {  // one wave issues the tensor DMA (EXEC ignored by TDM)
      unsigned long ga = (unsigned long)(const void*)p;
      unsigned la = (unsigned)(unsigned long)(&S[0]);
      u32x4 g0;
      i32x8 g1;
      i32x4 g2 = {0, 0, 0, 0}, g3 = {0, 0, 0, 0};
      g0[0] = 1u;                                                   // count=1
      g0[1] = la;                                                   // lds_addr
      g0[2] = (unsigned)ga;                                         // global_addr lo
      g0[3] = (unsigned)((ga >> 32) & 0x01FFFFFFu) | 0x80000000u;   // addr[56:32] | type=2
      int nd = FS * 2;                                              // 768 dwords (data_size=4B)
      g1[0] = (int)(2u << 16);                  // wg_mask=0, data_size=2 -> 4B
      g1[1] = (int)((unsigned)nd << 16);        // tensor_dim0[15:0]
      g1[2] = (int)(1u << 16);                  // tensor_dim0 hi, tensor_dim1=1
      g1[3] = (int)((unsigned)nd << 16);        // tile_dim0=768
      g1[4] = 1;                                // tile_dim1=1, tile_dim2=0
      g1[5] = nd;                               // tensor_dim0_stride
      g1[6] = 0;
      g1[7] = 0;
#if __clang_major__ >= 23
      i32x8 g4 = {0, 0, 0, 0, 0, 0, 0, 0};
      __builtin_amdgcn_tensor_load_to_lds(g0, g1, g2, g3, g4, 0);
#else
      __builtin_amdgcn_tensor_load_to_lds(g0, g1, g2, g3, 0);
#endif
      __builtin_amdgcn_s_wait_tensorcnt(0);
    }
    staged = true;
  }
#endif
  if (!staged)
    for (int i = tid; i < FS; i += 128) S[i] = p[(long)i * elemStride];
  __syncthreads();

  // decimate x[3m+r] into bit-reversed order per sub-FFT
  for (int i = tid; i < FS; i += 128) {
    int r = i % 3, m = i / 3;
    int mb = (int)(__brev((unsigned)m) >> 25);  // 7-bit reverse
    B[r * 128 + mb] = S[i];
  }
  __syncthreads();

  float sgn = inverse ? 1.0f : -1.0f;
  for (int len = 2; len <= 128; len <<= 1) {
    int half = len >> 1;
    for (int idx = tid; idx < 192; idx += 128) {
      int r = idx >> 6, j = idx & 63;
      int gpos = (j / half) * len + (j % half);
      int bi = r * 128 + gpos;
      float ang = sgn * 6.28318530717958647f * (float)(j % half) / (float)len;
      float s, c;
      __sincosf(ang, &s, &c);
      float2 u = B[bi], v = B[bi + half];
      float2 t = make_float2(c * v.x - s * v.y, c * v.y + s * v.x);
      B[bi] = make_float2(u.x + t.x, u.y + t.y);
      B[bi + half] = make_float2(u.x - t.x, u.y - t.y);
    }
    __syncthreads();
  }

  float nrm = inverse ? (1.0f / (float)FS) : 1.0f;
  for (int k = tid; k < FS; k += 128) {
    int km = k & 127;
    float2 acc = make_float2(0.f, 0.f);
#pragma unroll
    for (int r = 0; r < 3; ++r) {
      float ang = sgn * 6.28318530717958647f * (float)((r * k) % FS) / (float)FS;
      float s, c;
      __sincosf(ang, &s, &c);
      float2 v = B[r * 128 + km];
      acc.x += c * v.x - s * v.y;
      acc.y += c * v.y + s * v.x;
    }
    p[(long)k * elemStride] = make_float2(acc.x * nrm, acc.y * nrm);
  }
}

// =====================================================================
// Pointwise / setup kernels
// =====================================================================
__global__ void pad_to_cplx(const float* __restrict__ src, float2* __restrict__ dst, int Cc,
                            int shift) {
  GS_LOOP(i, (long)NB * Cc * FS2) {
    long pl = i / FS2;
    int f = (int)(i % FS2);
    int y = f / FS, x = f % FS;
    int sy = y - shift, sx = x - shift;
    float v = 0.f;
    if ((unsigned)sy < (unsigned)HV && (unsigned)sx < (unsigned)WV)
      v = src[(pl * HV + sy) * WV + sx];
    dst[i] = make_float2(v, 0.f);
  }
}

// z0 = soft_threshold(circ_shift(pad(fy9), (22,22)), b0) written as complex
__global__ void init_z(const float* __restrict__ fy9, const float* __restrict__ b0,
                       float2* __restrict__ Fz) {
  GS_LOOP(i, (long)NB * CH * FS2) {
    long pl = i / FS2;
    int f = (int)(i % FS2);
    int c = (int)(pl % CH);
    int y = f / FS, x = f % FS;
    int sy = (y + HK / 2) % FS, sx = (x + HK / 2) % FS;
    int py = sy - (HK - 1), px = sx - (HK - 1);
    float v = 0.f;
    if ((unsigned)py < (unsigned)HV && (unsigned)px < (unsigned)WV)
      v = fy9[(pl * HV + py) * WV + px];
    Fz[i] = make_float2(softt(v, b0[c]), 0.f);
  }
}

__global__ void init_delta(float2* __restrict__ kD) {
  GS_LOOP(i, (long)NB * FS2) {
    int f = (int)(i % FS2);
    int y = f / FS, x = f % FS;
    kD[i] = make_float2((y == HK / 2 && x == HK / 2) ? 1.f : 0.f, 0.f);
  }
}

// Fg = (zeta*conj(Fk)*Ffy + Fz) / (zeta*|Fk|^2 + 1 + eps)
__global__ void pw_fg(const float2* __restrict__ Ffy, const float2* __restrict__ Fk,
                      const float2* __restrict__ Fz, float2* __restrict__ Fg,
                      const float* __restrict__ zrow) {
  GS_LOOP(i, (long)NB * CH * FS2) {
    long n = i / ((long)CH * FS2);
    int c = (int)((i / FS2) % CH);
    int f = (int)(i % FS2);
    float2 fk = Fk[n * FS2 + f];
    float zeta = 10.f * zrow[c];
    float2 num = cadd(cscale(cmulcj(fk, Ffy[i]), zeta), Fz[i]);
    float den = zeta * mag2(fk) + 1.f + 1e-8f;
    Fg[i] = cscale(num, 1.f / den);
  }
}

// Z = soft_threshold(Re(G), b[c]) as complex
__global__ void thresh_cplx(const float2* __restrict__ G, const float* __restrict__ brow,
                            float2* __restrict__ Z) {
  GS_LOOP(i, (long)NB * CH * FS2) {
    int c = (int)((i / FS2) % CH);
    Z[i] = make_float2(softt(G[i].x, brow[c]), 0.f);
  }
}

// kC = (zk * sum_c conj(Fz)*Ffy + Fk) / (zk * sum_c |Fz|^2 + 1 + eps)
__global__ void kupdate(const float2* __restrict__ Fz, const float2* __restrict__ Ffy,
                        const float2* __restrict__ Fk, float2* __restrict__ kC,
                        const float* __restrict__ zkp) {
  float zk = zkp[0];
  GS_LOOP(i, (long)NB * FS2) {
    long n = i / FS2;
    int f = (int)(i % FS2);
    float2 s1 = make_float2(0.f, 0.f);
    float s2 = 0.f;
    for (int c = 0; c < CH; ++c) {
      long j = (n * CH + c) * (long)FS2 + f;
      float2 fz = Fz[j];
      s1 = cadd(s1, cmulcj(fz, Ffy[j]));
      s2 += mag2(fz);
    }
    float2 num = cadd(cscale(s1, zk), Fk[i]);
    float den = zk * s2 + 1.f + 1e-8f;
    kC[i] = cscale(num, 1.f / den);
  }
}

// per-batch: lse = logsumexp(k*100)/100, s45 = sum relu(k[:45,:45]-bk*lse)
__global__ __launch_bounds__(256) void k_reduce(const float2* __restrict__ kC,
                                                float* __restrict__ stats,
                                                const float* __restrict__ kb, int layer) {
  __shared__ float red[256];
  int nb = blockIdx.x, tid = threadIdx.x;
  const float2* src = kC + (long)nb * FS2;
  float mx = -1e30f;
  for (int i = tid; i < FS2; i += 256) mx = fmaxf(mx, src[i].x);
  red[tid] = mx;
  __syncthreads();
  for (int s = 128; s > 0; s >>= 1) {
    if (tid < s) red[tid] = fmaxf(red[tid], red[tid + s]);
    __syncthreads();
  }
  float m = red[0];
  __syncthreads();
  float se = 0.f;
  for (int i = tid; i < FS2; i += 256) se += __expf((src[i].x - m) * 100.f);
  red[tid] = se;
  __syncthreads();
  for (int s = 128; s > 0; s >>= 1) {
    if (tid < s) red[tid] += red[tid + s];
    __syncthreads();
  }
  float lse = m + logf(red[0]) * 0.01f;
  __syncthreads();
  float th = 0.01f * kb[layer] * lse;
  float s45 = 0.f;
  for (int i = tid; i < HK * HK; i += 256) {
    int y = i / HK, x = i % HK;
    s45 += fmaxf(src[y * FS + x].x - th, 0.f);
  }
  red[tid] = s45;
  __syncthreads();
  for (int s = 128; s > 0; s >>= 1) {
    if (tid < s) red[tid] += red[tid + s];
    __syncthreads();
  }
  if (tid == 0) {
    stats[nb * 2] = th;
    stats[nb * 2 + 1] = red[0];
  }
}

__global__ void k_finalize(const float2* __restrict__ kC, const float* __restrict__ stats,
                           float2* __restrict__ kD) {
  GS_LOOP(i, (long)NB * FS2) {
    int nb = (int)(i / FS2);
    int f = (int)(i % FS2);
    int y = f / FS, x = f % FS;
    float v = 0.f;
    if (y < HK && x < HK) {
      float t = fmaxf(kC[i].x - stats[nb * 2], 0.f);
      float d = (y == HK / 2 && x == HK / 2) ? 1e-8f : 0.f;
      v = (t + d) / (stats[nb * 2 + 1] + 1e-8f);
    }
    kD[i] = make_float2(v, 0.f);
  }
}

// circ_shift(pad(w0_unreflected), (1,1)) as complex; planes (ci, co)
__global__ void w0shift(const float* __restrict__ w0c, float2* __restrict__ Fw0) {
  GS_LOOP(i, (long)CIN * CH * FS2) {
    long pl = i / FS2;
    int f = (int)(i % FS2);
    int y = f / FS, x = f % FS;
    int sy = (y + 1) % FS, sx = (x + 1) % FS;
    float v = 0.f;
    if (sy < 3 && sx < 3) {
      int ci = (int)(pl / CH), co = (int)(pl % CH);
      v = w0c[(co * 3 + ci) * 9 + sy * 3 + sx];
    }
    Fw0[i] = make_float2(v, 0.f);
  }
}

// 3x3 complex Wiener solve per frequency
__global__ void wiener(const float2* __restrict__ Fk, const float2* __restrict__ Fy,
                       const float2* __restrict__ Fw0, const float2* __restrict__ Fg,
                       const float* __restrict__ eta, float2* __restrict__ Fx) {
  GS_LOOP(i, (long)NB * FS2) {
    long n = i / FS2;
    int f = (int)(i % FS2);
    float2 fk = Fk[i];
    float fk2 = mag2(fk);
    float srr = 0.f, sgg = 0.f, sbb = 0.f;
    float2 crg = make_float2(0, 0), crb = make_float2(0, 0), cgb = make_float2(0, 0);
    float2 br = make_float2(0, 0), bg = make_float2(0, 0), bbv = make_float2(0, 0);
    for (int c = 0; c < CH; ++c) {
      float e = 10.f * eta[c];
      float2 wr = Fw0[(0L * CH + c) * FS2 + f];
      float2 wg = Fw0[(1L * CH + c) * FS2 + f];
      float2 wb = Fw0[(2L * CH + c) * FS2 + f];
      float2 g = Fg[(n * CH + c) * (long)FS2 + f];
      srr += e * mag2(wr);
      sgg += e * mag2(wg);
      sbb += e * mag2(wb);
      crg = cadd(crg, cscale(cmulcj(wr, wg), e));
      crb = cadd(crb, cscale(cmulcj(wr, wb), e));
      cgb = cadd(cgb, cscale(cmulcj(wg, wb), e));
      br = cadd(br, cscale(cmulcj(wr, g), e));
      bg = cadd(bg, cscale(cmulcj(wg, g), e));
      bbv = cadd(bbv, cscale(cmulcj(wb, g), e));
    }
    float Crr = fk2 + srr, Cgg = fk2 + sgg, Cbb = fk2 + sbb;
    float2 fyr = Fy[(n * 3 + 0) * (long)FS2 + f];
    float2 fyg = Fy[(n * 3 + 1) * (long)FS2 + f];
    float2 fyb = Fy[(n * 3 + 2) * (long)FS2 + f];
    float2 Br = cadd(cmulcj(fk, fyr), br);
    float2 Bg = cadd(cmulcj(fk, fyg), bg);
    float2 Bb = cadd(cmulcj(fk, fyb), bbv);
    float crg2 = mag2(crg), crb2 = mag2(crb), cgb2 = mag2(cgb);
    float Irr = Cgg * Cbb - cgb2;
    float Igg = Crr * Cbb - crb2;
    float Ibb = Crr * Cgg - crg2;
    float2 Irg = csub(cmulcj(cgb, crb), cscale(crg, Cbb));
    float2 Irb = csub(cmul(crg, cgb), cscale(crb, Cgg));
    float2 Igb = csub(cmulcj(crg, crb), cscale(cgb, Crr));
    float2 t3 = cmul(crg, cgb);
    float re3 = t3.x * crb.x + t3.y * crb.y;  // Re(crg*cgb*conj(crb))
    float den = Crr * Irr - Cgg * crb2 - Cbb * crg2 + 2.f * re3 + 1e-8f;
    float inv = 1.f / den;
    float2 xr = cscale(cadd(cadd(cscale(Br, Irr), cmul(Irg, Bg)), cmul(Irb, Bb)), inv);
    float2 xg = cscale(cadd(cadd(cmulcj(Irg, Br), cscale(Bg, Igg)), cmul(Igb, Bb)), inv);
    float2 xb = cscale(cadd(cadd(cmulcj(Irb, Br), cmulcj(Igb, Bg)), cscale(Bb, Ibb)), inv);
    Fx[(n * 3 + 0) * (long)FS2 + f] = xr;
    Fx[(n * 3 + 1) * (long)FS2 + f] = xg;
    Fx[(n * 3 + 2) * (long)FS2 + f] = xb;
  }
}

__global__ void crop_out(const float2* __restrict__ X, float* __restrict__ out) {
  GS_LOOP(i, (long)NB * 3 * HV * WV) {
    long pl = i / (HV * WV);
    int rem = (int)(i % (HV * WV));
    int y = rem / WV, x = rem % WV;
    out[i] = X[pl * FS2 + (long)(y + HK / 2) * FS + (x + HK / 2)].x;
  }
}

__global__ void emit_k(const float2* __restrict__ kD, float* __restrict__ out) {
  GS_LOOP(i, (long)NB * HK * HK) {
    long n = i / (HK * HK);
    int rem = (int)(i % (HK * HK));
    int y = rem / HK, x = rem % HK;
    out[i] = kD[n * FS2 + (long)y * FS + x].x;
  }
}

// =====================================================================
// Host orchestration
// =====================================================================
static void fft2pass(float2* buf, int planes, int inverse, hipStream_t st) {
  int lines = planes * FS;
  // rows (contiguous -> TDM staging), then columns (strided)
  fft384_kernel<<<dim3(lines), dim3(128), 0, st>>>(buf, lines, FS, (long)FS2, (long)FS, 1L,
                                                   inverse);
  fft384_kernel<<<dim3(lines), dim3(128), 0, st>>>(buf, lines, FS, (long)FS2, 1L, (long)FS,
                                                   inverse);
}

extern "C" void kernel_launch(void* const* d_in, const int* in_sizes, int n_in, void* d_out,
                              int out_size, void* d_ws, size_t ws_size, hipStream_t stream) {
  (void)in_sizes; (void)n_in; (void)out_size; (void)ws_size;
  const float* blurred = (const float*)d_in[0];
  const float* w0 = (const float*)d_in[1];
  const float* wsW = (const float*)d_in[2];
  const float* biases = (const float*)d_in[3];
  const float* kb = (const float*)d_in[4];
  const float* kp = (const float*)d_in[5];
  const float* zetas = (const float*)d_in[6];
  const float* eta = (const float*)d_in[7];

  char* base = (char*)d_ws;
  size_t off = 0;
  auto alloc = [&](size_t bytes) -> void* {
    void* p = base + off;
    off = (off + bytes + 255) & ~(size_t)255;
    return p;
  };

  const long IMG = (long)NB * CH * HV * WV;
  const long CPLX = (long)NB * CH * FS2;
  _Float16* w0rH = (_Float16*)alloc(16 * 32 * sizeof(_Float16));
  _Float16* wsH = (_Float16*)alloc((size_t)(NL - 1) * CH * 160 * sizeof(_Float16));
  float* w0c = (float*)alloc(48 * 9 * sizeof(float));
  float* fyS = (float*)alloc((size_t)NL * IMG * sizeof(float));
  float2* Ffy = (float2*)alloc((size_t)CPLX * sizeof(float2));
  float2* Fz = (float2*)alloc((size_t)CPLX * sizeof(float2));
  float2* Fg = (float2*)alloc((size_t)CPLX * sizeof(float2));
  float2* Gt = (float2*)alloc((size_t)CPLX * sizeof(float2));
  float2* kD = (float2*)alloc((size_t)NB * FS2 * sizeof(float2));
  float2* kF = (float2*)alloc((size_t)NB * FS2 * sizeof(float2));
  float2* kC = (float2*)alloc((size_t)NB * FS2 * sizeof(float2));
  float2* FyB = (float2*)alloc((size_t)NB * 3 * FS2 * sizeof(float2));
  float2* Fw0B = (float2*)alloc((size_t)CIN * CH * FS2 * sizeof(float2));
  float2* FxB = (float2*)alloc((size_t)NB * 3 * FS2 * sizeof(float2));
  float* stats = (float*)alloc(NB * 2 * sizeof(float));

  // ---- weight prep ----
  prep_w0<<<dim3(1), dim3(64), 0, stream>>>(w0, w0rH, w0c);
  prep_ws<<<dim3(64), dim3(256), 0, stream>>>(wsW, wsH);

  // ---- forward conv chain (WMMA) ----
  dim3 cgrid(WV / 128, HV, NB);
  conv3x3_wmma<CIN, 32><<<cgrid, dim3(256), 0, stream>>>(blurred, w0rH, fyS);
  for (int l = 1; l < NL; ++l)
    conv3x3_wmma<CH, 160><<<cgrid, dim3(256), 0, stream>>>(fyS + (long)(l - 1) * IMG,
                                                           wsH + (long)(l - 1) * CH * 160,
                                                           fyS + (long)l * IMG);

  // ---- initial z, delta kernel ----
  init_z<<<dim3(2048), dim3(256), 0, stream>>>(fyS + (long)(NL - 1) * IMG, biases, Fz);
  fft2pass(Fz, NB * CH, 0, stream);
  init_delta<<<dim3(512), dim3(256), 0, stream>>>(kD);

  // ---- prox iterations ----
  for (int layer = 0; layer < NL; ++layer) {
    pad_to_cplx<<<dim3(2048), dim3(256), 0, stream>>>(fyS + (long)(NL - 1 - layer) * IMG, Ffy,
                                                      CH, HK - 1);
    fft2pass(Ffy, NB * CH, 0, stream);
    hipMemcpyAsync(kF, kD, (size_t)NB * FS2 * sizeof(float2), hipMemcpyDeviceToDevice, stream);
    fft2pass(kF, NB, 0, stream);
    pw_fg<<<dim3(2048), dim3(256), 0, stream>>>(Ffy, kF, Fz, Fg, zetas + (long)layer * CH);
    hipMemcpyAsync(Gt, Fg, (size_t)CPLX * sizeof(float2), hipMemcpyDeviceToDevice, stream);
    fft2pass(Gt, NB * CH, 1, stream);
    thresh_cplx<<<dim3(2048), dim3(256), 0, stream>>>(Gt, biases + (long)(layer + 1) * CH, Fz);
    fft2pass(Fz, NB * CH, 0, stream);
    kupdate<<<dim3(1024), dim3(256), 0, stream>>>(Fz, Ffy, kF, kC, kp + layer);
    fft2pass(kC, NB, 1, stream);
    k_reduce<<<dim3(NB), dim3(256), 0, stream>>>(kC, stats, kb, layer);
    k_finalize<<<dim3(512), dim3(256), 0, stream>>>(kC, stats, kD);
  }

  // ---- final Wiener solve ----
  pad_to_cplx<<<dim3(2048), dim3(256), 0, stream>>>(blurred, FyB, CIN, HK - 1);
  fft2pass(FyB, NB * CIN, 0, stream);
  hipMemcpyAsync(kF, kD, (size_t)NB * FS2 * sizeof(float2), hipMemcpyDeviceToDevice, stream);
  fft2pass(kF, NB, 0, stream);
  w0shift<<<dim3(2048), dim3(256), 0, stream>>>(w0c, Fw0B);
  fft2pass(Fw0B, CIN * CH, 0, stream);
  hipMemcpyAsync(Gt, Fg, (size_t)CPLX * sizeof(float2), hipMemcpyDeviceToDevice, stream);
  fft2pass(Gt, NB * CH, 1, stream);
  thresh_cplx<<<dim3(2048), dim3(256), 0, stream>>>(Gt, biases + (long)(NL + 1) * CH, Fg);
  fft2pass(Fg, NB * CH, 0, stream);
  wiener<<<dim3(1024), dim3(256), 0, stream>>>(kF, FyB, Fw0B, Fg, eta, FxB);
  fft2pass(FxB, NB * CIN, 1, stream);
  crop_out<<<dim3(2048), dim3(256), 0, stream>>>(FxB, (float*)d_out);
  emit_k<<<dim3(64), dim3(256), 0, stream>>>(kD, (float*)d_out + (long)NB * 3 * HV * WV);
}
